// AttnDecoder_12610023981077
// MI455X (gfx1250) — compile-verified, matrix operand order checked by reading, not run on hardware
//
#include <hip/hip_runtime.h>
#include <hip/hip_bf16.h>

#define BB 64
#define SS 256
#define HH 1024
#define OO 64

typedef __attribute__((ext_vector_type(16))) __bf16 v16bf;
typedef __attribute__((ext_vector_type(8)))  float  v8f;

// ---------------------------------------------------------------------------
// WMMA helpers (CDNA5: D = A(16x32 bf16) * B(32x16 bf16) + C(16x16 f32))
// ---------------------------------------------------------------------------
static __device__ __forceinline__ v8f wmma_bf16(v16bf a, v16bf b, v8f c) {
  return __builtin_amdgcn_wmma_f32_16x16x32_bf16(
      /*neg_a=*/false, a, /*neg_b=*/false, b,
      /*c_mod=*/(short)0, c, /*reuse_a=*/false, /*reuse_b=*/false);
}

// Load one 16x32-bf16 fragment row (A: row m of A; B: row n of W for x@W^T).
// Per ISA layout: lane holds K = base+{0..7} in VGPR0..3 and base+16+{0..7}
// in VGPR4..7, base = (lane>=16) ? 8 : 0.  p must point at (row, k0+base).
static __device__ __forceinline__ v16bf load_frag(const __hip_bfloat16* p) {
  union { uint4 u[2]; v16bf v; } x;
  x.u[0] = *reinterpret_cast<const uint4*>(p);
  x.u[1] = *reinterpret_cast<const uint4*>(p + 16);
  return x.v;
}

static __device__ __forceinline__ float fast_sigmoid(float x) {
  return 1.f / (1.f + __expf(-x));
}
static __device__ __forceinline__ float fast_tanh(float x) {
  x = fminf(fmaxf(x, -15.f), 15.f);
  float e = __expf(2.f * x);
  return (e - 1.f) / (e + 1.f);
}

// ---------------------------------------------------------------------------
// fp32 -> bf16 convert
// ---------------------------------------------------------------------------
__global__ void cvt_bf16_kernel(const float* __restrict__ src,
                                __hip_bfloat16* __restrict__ dst, int n) {
  int i = blockIdx.x * blockDim.x + threadIdx.x;
  if (i < n) dst[i] = __float2bfloat16(src[i]);
}

// ---------------------------------------------------------------------------
// x0 = encoder_inputs[:, -1, :] @ embed_W^T + embed_b  -> xcat[:, 0:H] (bf16)
// ---------------------------------------------------------------------------
__global__ void embed_kernel(const float* __restrict__ enc_in,
                             const float* __restrict__ embW,
                             const float* __restrict__ embB,
                             __hip_bfloat16* __restrict__ xcat) {
  int i = blockIdx.x * blockDim.x + threadIdx.x;   // 0..B*H
  int b = i >> 10, h = i & (HH - 1);
  const float* x = enc_in + ((size_t)b * SS + (SS - 1)) * OO;
  const float* w = embW + (size_t)h * OO;
  float acc = embB[h];
#pragma unroll 8
  for (int o = 0; o < OO; ++o) acc += x[o] * w[o];
  xcat[(size_t)b * (2 * HH) + h] = __float2bfloat16(acc);
}

// ---------------------------------------------------------------------------
// Generic WMMA GEMM: out[M,N] = A[M,K] @ W[N,K]^T + bias[n]
// One wave computes a 64x16 strip (4 M-tiles reuse each B fragment).
// ---------------------------------------------------------------------------
template <bool OUT_BF16>
__global__ void gemm_bf16_kernel(const __hip_bfloat16* __restrict__ A,
                                 const __hip_bfloat16* __restrict__ W,
                                 const float* __restrict__ bias,
                                 void* __restrict__ out,
                                 int M, int N, int K) {
  int wid  = (blockIdx.x * blockDim.x + threadIdx.x) >> 5;
  int lane = threadIdx.x & 31;
  int tilesN = N >> 4;
  int m0 = (wid / tilesN) << 6;
  int n0 = (wid % tilesN) << 4;
  if (m0 >= M) return;                      // uniform per-wave
  int half = lane >> 4, lr = lane & 15, kb = half * 8;
  int n = n0 + lr;
  float bv = bias ? bias[n] : 0.f;
  v8f acc[4];
#pragma unroll
  for (int t = 0; t < 4; ++t)
#pragma unroll
    for (int v = 0; v < 8; ++v) acc[t][v] = bv;

  const __hip_bfloat16* wr = W + (size_t)n * K + kb;
  for (int k = 0; k < K; k += 32) {
    v16bf bf = load_frag(wr + k);
#pragma unroll
    for (int t = 0; t < 4; ++t) {
      v16bf a = load_frag(A + (size_t)(m0 + t * 16 + lr) * K + kb + k);
      acc[t] = wmma_bf16(a, bf, acc[t]);
    }
  }
#pragma unroll
  for (int t = 0; t < 4; ++t)
#pragma unroll
    for (int v = 0; v < 8; ++v) {
      int m = m0 + t * 16 + v + half * 8;   // C layout: VGPR v, half -> M
      float val = acc[t][v];
      if (OUT_BF16)
        ((__hip_bfloat16*)out)[(size_t)m * N + n] = __float2bfloat16(val);
      else
        ((float*)out)[(size_t)m * N + n] = val;
    }
}

// ---------------------------------------------------------------------------
// Attention step: scores = Va . tanh(Waq[b] + Uk[b,s]) ; softmax ; ctx.
// One block per batch element, 1024 threads:
//   score phase : 4 threads per source position, each reduces 256 h (packed
//                 bf16x2 loads), combined through LDS.
//   softmax     : threads 0..255 (tree reduce in LDS).
//   ctx phase   : one h per thread, 256-iter weighted sum over s.
// ---------------------------------------------------------------------------
__global__ __launch_bounds__(1024)
void attn_step_kernel(const __hip_bfloat16* __restrict__ Waq,
                      const __hip_bfloat16* __restrict__ Uk,
                      const __hip_bfloat16* __restrict__ enc_bf,
                      const float* __restrict__ Va,
                      const float* __restrict__ Vab,
                      float* __restrict__ attn_out,
                      __hip_bfloat16* __restrict__ xcat, int t) {
  __shared__ float part[1024];
  __shared__ float sc[SS];
  __shared__ float red[SS];
  const int b = blockIdx.x;
  const int tid = threadIdx.x;
  const int s = tid >> 2;        // source position 0..255
  const int j = tid & 3;         // h-chunk 0..3 (256 h each)

  // ---- partial scores ----
  {
    const __hip_bfloat162* uk2 =
        reinterpret_cast<const __hip_bfloat162*>(Uk + ((size_t)b * SS + s) * HH + j * 256);
    const __hip_bfloat162* wq2 =
        reinterpret_cast<const __hip_bfloat162*>(Waq + (size_t)b * HH + j * 256);
    const float2* va2 = reinterpret_cast<const float2*>(Va + j * 256);
    float p = 0.f;
#pragma unroll 4
    for (int h = 0; h < 128; ++h) {
      float2 q = __bfloat1622float2(wq2[h]);
      float2 u = __bfloat1622float2(uk2[h]);
      float2 v = va2[h];
      p = fmaf(v.x, fast_tanh(q.x + u.x), p);
      p = fmaf(v.y, fast_tanh(q.y + u.y), p);
    }
    part[tid] = p;
  }
  __syncthreads();

  // ---- combine partials + softmax over 256 source positions ----
  float score = 0.f;
  if (tid < SS) {
    score = part[4 * tid] + part[4 * tid + 1] + part[4 * tid + 2] +
            part[4 * tid + 3] + Vab[0];
    red[tid] = score;
  }
  __syncthreads();
  for (int off = 128; off; off >>= 1) {
    if (tid < off) red[tid] = fmaxf(red[tid], red[tid + off]);
    __syncthreads();
  }
  float mx = red[0];
  __syncthreads();
  if (tid < SS) {
    float e = __expf(score - mx);
    sc[tid] = e;
    red[tid] = e;
  }
  __syncthreads();
  for (int off = 128; off; off >>= 1) {
    if (tid < off) red[tid] += red[tid + off];
    __syncthreads();
  }
  float inv = 1.f / red[0];
  __syncthreads();
  if (tid < SS) {
    float w = sc[tid] * inv;
    sc[tid] = w;
    attn_out[((size_t)b * SS + t) * SS + tid] = w;
  }
  __syncthreads();

  // ---- ctx[b,h] = sum_s w[s] * enc[b,s,h]  (h == tid, coalesced) ----
  {
    const __hip_bfloat16* ep = enc_bf + (size_t)b * SS * HH + tid;
    float c = 0.f;
#pragma unroll 4
    for (int s2 = 0; s2 < SS; ++s2)
      c = fmaf(sc[s2], __bfloat162float(ep[(size_t)s2 * HH]), c);
    xcat[(size_t)b * (2 * HH) + HH + tid] = __float2bfloat16(c);
  }
}

// ---------------------------------------------------------------------------
// One GRU layer step.  Each wave -> one 16x16 tile of h'.
// acc_r/acc_z accumulate (x@Wih^T + h@Whh^T + both biases) directly in C.
// ---------------------------------------------------------------------------
__global__ void gru_step_kernel(const __hip_bfloat16* __restrict__ x_bf, int Kx,
                                const __hip_bfloat16* __restrict__ h_bf_old,
                                const float* __restrict__ h_fp_old,
                                const __hip_bfloat16* __restrict__ Wih,
                                const __hip_bfloat16* __restrict__ Whh,
                                const float* __restrict__ bih,
                                const float* __restrict__ bhh,
                                float* __restrict__ h_fp_new,
                                __hip_bfloat16* __restrict__ h_bf_new,
                                __hip_bfloat16* __restrict__ outs_bf,  // may be null
                                __hip_bfloat16* __restrict__ xfb,      // may be null
                                int t) {
  int wid  = (blockIdx.x * blockDim.x + threadIdx.x) >> 5;
  int lane = threadIdx.x & 31;
  int n0 = (wid & 63) << 4;            // H/16 = 64 column tiles
  int m0 = (wid >> 6) << 4;            // B/16 = 4 row tiles
  int half = lane >> 4, lr = lane & 15, kb = half * 8;
  int n = n0 + lr;

  v8f aR, aZ, aIN, aHN;
  float bR  = bih[n] + bhh[n];
  float bZ  = bih[HH + n] + bhh[HH + n];
  float bIN = bih[2 * HH + n];
  float bHN = bhh[2 * HH + n];
#pragma unroll
  for (int v = 0; v < 8; ++v) { aR[v] = bR; aZ[v] = bZ; aIN[v] = bIN; aHN[v] = bHN; }

  const __hip_bfloat16* ar = x_bf + (size_t)(m0 + lr) * Kx + kb;
  const __hip_bfloat16* wr = Wih + (size_t)n * Kx + kb;
  const __hip_bfloat16* wz = Wih + (size_t)(HH + n) * Kx + kb;
  const __hip_bfloat16* wn = Wih + (size_t)(2 * HH + n) * Kx + kb;
  for (int k = 0; k < Kx; k += 32) {
    v16bf a = load_frag(ar + k);
    aR  = wmma_bf16(a, load_frag(wr + k), aR);
    aZ  = wmma_bf16(a, load_frag(wz + k), aZ);
    aIN = wmma_bf16(a, load_frag(wn + k), aIN);
  }
  const __hip_bfloat16* hr = h_bf_old + (size_t)(m0 + lr) * HH + kb;
  const __hip_bfloat16* ur = Whh + (size_t)n * HH + kb;
  const __hip_bfloat16* uz = Whh + (size_t)(HH + n) * HH + kb;
  const __hip_bfloat16* un = Whh + (size_t)(2 * HH + n) * HH + kb;
  for (int k = 0; k < HH; k += 32) {
    v16bf a = load_frag(hr + k);
    aR  = wmma_bf16(a, load_frag(ur + k), aR);
    aZ  = wmma_bf16(a, load_frag(uz + k), aZ);
    aHN = wmma_bf16(a, load_frag(un + k), aHN);
  }
#pragma unroll
  for (int v = 0; v < 8; ++v) {
    int m = m0 + v + half * 8;           // batch index
    float r  = fast_sigmoid(aR[v]);
    float z  = fast_sigmoid(aZ[v]);
    float nn = fast_tanh(aIN[v] + r * aHN[v]);
    float hnew = (1.f - z) * nn + z * h_fp_old[(size_t)m * HH + n];
    h_fp_new[(size_t)m * HH + n] = hnew;
    __hip_bfloat16 hb = __float2bfloat16(hnew);
    h_bf_new[(size_t)m * HH + n] = hb;
    if (outs_bf) outs_bf[((size_t)t * BB + m) * HH + n] = hb;
    if (xfb)     xfb[(size_t)m * (2 * HH) + n] = hb;    // feedback x_in
  }
}

// ---------------------------------------------------------------------------
// dec[b,t,o] = outs[t,b,:] @ out_W[o,:]^T + out_b[o]
// ---------------------------------------------------------------------------
__global__ void dec_gemm_kernel(const __hip_bfloat16* __restrict__ outs_bf,
                                const __hip_bfloat16* __restrict__ outW,
                                const float* __restrict__ outB,
                                float* __restrict__ dec) {
  int wid  = (blockIdx.x * blockDim.x + threadIdx.x) >> 5;
  int lane = threadIdx.x & 31;
  const int tilesN = OO >> 4;                  // 4
  int m0 = (wid / tilesN) << 6;                // rows = t*B + b
  int n0 = (wid % tilesN) << 4;
  int half = lane >> 4, lr = lane & 15, kb = half * 8;
  int n = n0 + lr;
  float bv = outB[n];
  v8f acc[4];
#pragma unroll
  for (int t = 0; t < 4; ++t)
#pragma unroll
    for (int v = 0; v < 8; ++v) acc[t][v] = bv;

  const __hip_bfloat16* wr = outW + (size_t)n * HH + kb;
  for (int k = 0; k < HH; k += 32) {
    v16bf bf = load_frag(wr + k);
#pragma unroll
    for (int t = 0; t < 4; ++t) {
      v16bf a = load_frag(outs_bf + (size_t)(m0 + t * 16 + lr) * HH + kb + k);
      acc[t] = wmma_bf16(a, bf, acc[t]);
    }
  }
#pragma unroll
  for (int t = 0; t < 4; ++t)
#pragma unroll
    for (int v = 0; v < 8; ++v) {
      int m = m0 + t * 16 + v + half * 8;
      int ts = m >> 6, b = m & 63;
      dec[(size_t)b * SS * OO + (size_t)ts * OO + n] = acc[t][v];
    }
}

// ---------------------------------------------------------------------------
// Host launch
// ---------------------------------------------------------------------------
extern "C" void kernel_launch(void* const* d_in, const int* in_sizes, int n_in,
                              void* d_out, int out_size, void* d_ws, size_t ws_size,
                              hipStream_t stream) {
  const float* enc_in  = (const float*)d_in[0];
  const float* enc_out = (const float*)d_in[1];
  const float* enc_hid = (const float*)d_in[2];
  const float* embW    = (const float*)d_in[3];
  const float* embB    = (const float*)d_in[4];
  const float* WaW     = (const float*)d_in[5];
  const float* WaB     = (const float*)d_in[6];
  const float* UaW     = (const float*)d_in[7];
  const float* UaB     = (const float*)d_in[8];
  const float* VaW     = (const float*)d_in[9];
  const float* VaB     = (const float*)d_in[10];
  const float* Wih0    = (const float*)d_in[11];
  const float* Whh0    = (const float*)d_in[12];
  const float* bih0    = (const float*)d_in[13];
  const float* bhh0    = (const float*)d_in[14];
  const float* Wih1    = (const float*)d_in[15];
  const float* Whh1    = (const float*)d_in[16];
  const float* bih1    = (const float*)d_in[17];
  const float* bhh1    = (const float*)d_in[18];
  const float* outW    = (const float*)d_in[19];
  const float* outB    = (const float*)d_in[20];

  float* out      = (float*)d_out;
  float* dec_out  = out;                                   // [B,S,O]
  float* hfin_out = out + (size_t)BB * SS * OO;            // [2,B,H]
  float* attn_out = hfin_out + (size_t)2 * BB * HH;        // [B,S,S]

  // ---- workspace carve (bf16 weights/activations + fp32 state) ----
  char* wp = (char*)d_ws;
  auto carve = [&](size_t bytes) -> void* {
    void* p = wp; wp += (bytes + 255) & ~(size_t)255; return p;
  };
  typedef __hip_bfloat16 bf;
  bf* Wa_bf   = (bf*)carve((size_t)HH * HH * 2);
  bf* Ua_bf   = (bf*)carve((size_t)HH * HH * 2);
  bf* Wih0_bf = (bf*)carve((size_t)3 * HH * 2 * HH * 2);
  bf* Whh0_bf = (bf*)carve((size_t)3 * HH * HH * 2);
  bf* Wih1_bf = (bf*)carve((size_t)3 * HH * HH * 2);
  bf* Whh1_bf = (bf*)carve((size_t)3 * HH * HH * 2);
  bf* outW_bf = (bf*)carve((size_t)OO * HH * 2);
  bf* enc_bf  = (bf*)carve((size_t)BB * SS * HH * 2);
  bf* Uk_bf   = (bf*)carve((size_t)BB * SS * HH * 2);
  bf* outs_bf = (bf*)carve((size_t)SS * BB * HH * 2);
  bf* xcat_bf = (bf*)carve((size_t)BB * 2 * HH * 2);
  bf* Waq_bf  = (bf*)carve((size_t)BB * HH * 2);
  bf* h0_bf[2], *h1_bf[2];
  float* h0_fp[2]; float* h1_fp[2];
  for (int i = 0; i < 2; ++i) {
    h0_bf[i] = (bf*)carve((size_t)BB * HH * 2);
    h1_bf[i] = (bf*)carve((size_t)BB * HH * 2);
    h0_fp[i] = (float*)carve((size_t)BB * HH * 4);
    h1_fp[i] = (float*)carve((size_t)BB * HH * 4);
  }

  const int T = 256;  // threads per block (8 waves)
  auto cvt = [&](const float* s, bf* d, int n) {
    cvt_bf16_kernel<<<(n + T - 1) / T, T, 0, stream>>>(s, d, n);
  };

  // ---- one-time preprocessing ----
  cvt(WaW,  Wa_bf,   HH * HH);
  cvt(UaW,  Ua_bf,   HH * HH);
  cvt(Wih0, Wih0_bf, 3 * HH * 2 * HH);
  cvt(Whh0, Whh0_bf, 3 * HH * HH);
  cvt(Wih1, Wih1_bf, 3 * HH * HH);
  cvt(Whh1, Whh1_bf, 3 * HH * HH);
  cvt(outW, outW_bf, OO * HH);
  cvt(enc_out, enc_bf, BB * SS * HH);
  cvt(enc_hid,            h0_bf[0], BB * HH);
  cvt(enc_hid + BB * HH,  h1_bf[0], BB * HH);
  hipMemcpyAsync(h0_fp[0], enc_hid,           (size_t)BB * HH * 4, hipMemcpyDeviceToDevice, stream);
  hipMemcpyAsync(h1_fp[0], enc_hid + BB * HH, (size_t)BB * HH * 4, hipMemcpyDeviceToDevice, stream);

  embed_kernel<<<(BB * HH) / T, T, 0, stream>>>(enc_in, embW, embB, xcat_bf);

  // Uk = enc_out @ Ua^T + Ua_b   (M = B*S = 16384, N = K = 1024)
  {
    int waves = (BB * SS / 64) * (HH / 16);        // 16384
    gemm_bf16_kernel<true><<<waves / 8, T, 0, stream>>>(
        enc_bf, Ua_bf, UaB, (void*)Uk_bf, BB * SS, HH, HH);
  }

  // ---- 256 dependent decode steps ----
  for (int t = 0; t < SS; ++t) {
    int pi = t & 1, po = pi ^ 1;
    // Waq = h1 @ Wa^T + Wa_b  (M=64, N=K=1024) -> 64 waves
    gemm_bf16_kernel<true><<<8, T, 0, stream>>>(
        h1_bf[pi], Wa_bf, WaB, (void*)Waq_bf, BB, HH, HH);
    // attention scores + softmax + context (writes attn_out row t, xcat ctx)
    attn_step_kernel<<<BB, 1024, 0, stream>>>(
        Waq_bf, Uk_bf, enc_bf, VaW, VaB, attn_out, xcat_bf, t);
    // GRU layer 0: x = [x_in | ctx] (K=2H) -> 256 waves
    gru_step_kernel<<<32, T, 0, stream>>>(
        xcat_bf, 2 * HH, h0_bf[pi], h0_fp[pi], Wih0_bf, Whh0_bf, bih0, bhh0,
        h0_fp[po], h0_bf[po], nullptr, nullptr, t);
    // GRU layer 1: x = h0' (K=H); also writes outs[t] and x_in feedback
    gru_step_kernel<<<32, T, 0, stream>>>(
        h0_bf[po], HH, h1_bf[pi], h1_fp[pi], Wih1_bf, Whh1_bf, bih1, bhh1,
        h1_fp[po], h1_bf[po], outs_bf, xcat_bf, t);
  }

  // dec = outs @ out_W^T + out_b   (M = S*B = 16384, N = 64, K = 1024)
  {
    int waves = (SS * BB / 64) * (OO / 16);        // 1024
    dec_gemm_kernel<<<waves / 8, T, 0, stream>>>(outs_bf, outW_bf, outB, dec_out);
  }

  // h_final = [h0, h1] after last step (final write parity = 0 for even S)
  int fin = SS & 1;
  hipMemcpyAsync(hfin_out,           h0_fp[fin], (size_t)BB * HH * 4, hipMemcpyDeviceToDevice, stream);
  hipMemcpyAsync(hfin_out + BB * HH, h1_fp[fin], (size_t)BB * HH * 4, hipMemcpyDeviceToDevice, stream);
}
